// DynaMixerBlock_30416958390614
// MI455X (gfx1250) — matrix operand order repaired and checked by reading, not compiled, not run on hardware
//
#include <hip/hip_runtime.h>
#include <hip/hip_bf16.h>

typedef __attribute__((ext_vector_type(16))) _Float16 v16h;
typedef __attribute__((ext_vector_type(8)))  float    v8f;
typedef __attribute__((ext_vector_type(4)))  int      v4i;

#define DIMC 384
#define RESL 32
#define NHD  8
#define NBATCH 64
#define NTOK (NBATCH*RESL*RESL)   /* 65536 tokens */
#define NSEQ (NBATCH*RESL)        /* 2048 sequences per direction */

#if __has_builtin(__builtin_amdgcn_global_load_async_to_lds_b128) && \
    __has_builtin(__builtin_amdgcn_s_wait_asynccnt)
#define HAVE_ASYNC_LDS 1
#endif

union H16 { v16h v; _Float16 h[16]; uint4 q[2]; };

// D = A x B + C, 16x16x32 f16 -> f32
__device__ __forceinline__ v8f wmma32(v16h a, v16h b, v8f c) {
  return __builtin_amdgcn_wmma_f32_16x16x32_f16(false, a, false, b, (short)0, c,
                                                false, false);
}

// A tile [16 x 32] f16 row-major, leading dim ld (halves).
// Layout: lanes 0-15 row M=lane, elems0..15=K0..15;
// lanes 16-31 same rows, K16..31.
__device__ __forceinline__ v16h ldA(const _Float16* s, int ld) {
  const int lane = threadIdx.x & 31;
  const int row  = lane & 15;
  const int kof  = (lane >> 4) << 4;       // 0 or 16
  H16 u;
  u.q[0] = *(const uint4*)(s + row * ld + kof);
  u.q[1] = *(const uint4*)(s + row * ld + kof + 8);
  return u.v;
}

// B tile for X @ W^T where W is [N,K] row-major: B[k,n]=W[n,k].
// B layout used here (same k order as ldA): lane n holds K0..15 (lanes 0-15) or K16..31 (lanes 16-31)
// of column n -> 16 contiguous halves of W row (lane&15).
__device__ __forceinline__ v16h ldBt(const _Float16* s, int ld) {
  const int lane = threadIdx.x & 31;
  const int n    = lane & 15;
  const int kof  = (lane >> 4) << 4;       // 0 or 16
  H16 u;
  const uint4* p = (const uint4*)(s + n * ld + kof);
  u.q[0] = p[0];
  u.q[1] = p[1];
  return u.v;
}

// B tile from a genuinely [K,N] row-major f16 matrix (strided per element).
__device__ __forceinline__ v16h ldB(const _Float16* s, int ld) {
  const int lane = threadIdx.x & 31;
  const int n    = lane & 15;
  const int kof  = (lane >> 4) << 4;
  H16 u;
#pragma unroll
  for (int e = 0; e < 16; ++e) u.h[e] = s[(kof + e) * ld + n];
  return u.v;
}

// 16-byte global -> LDS copy; async (ASYNCcnt) when the toolchain has the
// gfx1250 builtin, else a plain VGPR round-trip.
__device__ __forceinline__ void cp16_g2l(const _Float16* g, _Float16* l) {
#ifdef HAVE_ASYNC_LDS
  __builtin_amdgcn_global_load_async_to_lds_b128(
      (__attribute__((address_space(1))) v4i*)(g),
      (__attribute__((address_space(3))) v4i*)(l), 0, 0);
#else
  *(uint4*)l = *(const uint4*)g;
#endif
}

__device__ __forceinline__ void cp_fence_lds() {
#ifdef HAVE_ASYNC_LDS
  __builtin_amdgcn_s_wait_asynccnt(0);
#endif
}

// ---------------------------------------------------------------- cast f32->f16
__global__ void cast_f32_f16_kernel(const float* __restrict__ src,
                                    _Float16* __restrict__ dst, int n) {
  int i = blockIdx.x * blockDim.x + threadIdx.x;
  if (i < n) dst[i] = (_Float16)src[i];
}

// ---------------------------------------------------------------- mixing kernel
// One block (256 thr = 8 waves) per sequence of L=32 tokens.
// dir==0: mix along H (seq = (b, wcol)); dir==1: mix along W (seq = (b, hrow)).
__global__ void __launch_bounds__(256)
mix_kernel(const float* __restrict__ x,
           const float* __restrict__ Wc, const float* __restrict__ bc,
           const _Float16* __restrict__ Wg16, const float* __restrict__ bg,
           _Float16* __restrict__ ymix, int dir) {
  __shared__ __align__(16) _Float16 xs[RESL][DIMC + 8];   // 32 x 392 f16
  __shared__ __align__(16) _Float16 w2[16][72];           // compressed, padded rows
  __shared__ __align__(16) _Float16 gsm[NHD][RESL][RESL]; // gen out -> softmaxed

  const int tid  = threadIdx.x;
  const int lane = tid & 31;
  const int wave = tid >> 5;
  const int s    = blockIdx.x;
  const int b    = s >> 5;
  const int q    = s & 31;

  // zero-pad compressed matrix (rows 8..15 must be 0 for the padded A tile)
  for (int i = tid; i < 16 * 72; i += 256) (&w2[0][0])[i] = (_Float16)0.f;

  // stage sequence to LDS, f32 -> f16
  for (int i = tid; i < RESL * (DIMC / 4); i += 256) {
    const int l  = i / (DIMC / 4);
    const int c4 = (i % (DIMC / 4)) * 4;
    const int tok = (dir == 0) ? ((b * RESL + l) * RESL + q)
                               : ((b * RESL + q) * RESL + l);
    const float4 f = *(const float4*)(x + (size_t)tok * DIMC + c4);
    xs[l][c4 + 0] = (_Float16)f.x;
    xs[l][c4 + 1] = (_Float16)f.y;
    xs[l][c4 + 2] = (_Float16)f.z;
    xs[l][c4 + 3] = (_Float16)f.w;
  }
  __syncthreads();

  // compress: w1[l,o] = x[l,:] . Wc[o,:] + bc[o]; repack to [NH, L*RD]
  for (int idx = tid; idx < RESL * 16; idx += 256) {
    const int l = idx >> 4, o = idx & 15;
    const float* wrow = Wc + o * DIMC;
    float sum = bc[o];
#pragma unroll 4
    for (int k = 0; k < DIMC; ++k) sum += (float)xs[l][k] * wrow[k];
    w2[o >> 1][(l << 1) | (o & 1)] = (_Float16)sum;  // [h][l*2+r]
  }
  __syncthreads();

  // generate: [16x64] @ Wg^T (1024x64) -> rows 0..7 valid; WMMA, K=64 = 2 steps
  {
    const v16h a0 = ldA(&w2[0][0], 72);
    const v16h a1 = ldA(&w2[0][32], 72);
    for (int t = 0; t < 8; ++t) {
      const int nt = wave * 8 + t;                      // 64 N-tiles of 1024
      const v16h b0 = ldBt(Wg16 + (size_t)(nt * 16) * 64, 64);
      const v16h b1 = ldBt(Wg16 + (size_t)(nt * 16) * 64 + 32, 64);
      v8f acc = {};
      acc = wmma32(a0, b0, acc);
      acc = wmma32(a1, b1, acc);
      if (lane < 16) {                                  // lanes 0-15 hold M=0..7
        const int n = nt * 16 + lane;
        const int l = n >> 5, j = n & 31;
        const float bias = bg[n];
#pragma unroll
        for (int v = 0; v < 8; ++v)
          gsm[v][l][j] = (_Float16)(acc[v] + bias);     // head = v
      }
    }
  }
  __syncthreads();

  // softmax over l (rows) for each (h, j): 256 columns, one per thread
  {
    const int h = tid >> 5, j = tid & 31;
    float col[RESL];
    float m = -1e30f;
#pragma unroll
    for (int l = 0; l < RESL; ++l) {
      col[l] = (float)gsm[h][l][j];
      m = fmaxf(m, col[l]);
    }
    float ssum = 0.f;
#pragma unroll
    for (int l = 0; l < RESL; ++l) { col[l] = __expf(col[l] - m); ssum += col[l]; }
    const float inv = 1.f / ssum;
#pragma unroll
    for (int l = 0; l < RESL; ++l) gsm[h][l][j] = (_Float16)(col[l] * inv);
  }
  __syncthreads();

  // mix: y[h,d,j] = sum_l x[l, h*48+d] * wsm[h,l,j].  One head per wave,
  // 3 M-tiles (48 ch) x 2 N-tiles, K=32 -> 6 WMMAs per wave.
  {
    const int head = wave;
    for (int i = 0; i < 6; ++i) {
      const int mt = i >> 1, nt = i & 1;
      // A[m=d, k=l] = xs[l][head*48 + d] (column reads from LDS)
      H16 ua;
      const int d   = mt * 16 + (lane & 15);
      const int cch = head * 48 + d;
      const int kof = (lane >> 4) << 4;
#pragma unroll
      for (int e = 0; e < 8; ++e) {
        ua.h[e]     = xs[kof + e][cch];
        ua.h[8 + e] = xs[kof + 8 + e][cch];
      }
      const v16h bm = ldB(&gsm[head][0][nt * 16], RESL);
      v8f acc = {};
      acc = wmma32(ua.v, bm, acc);
      // y.transpose -> token (l=j), channel head*48+d
      const int j   = nt * 16 + (lane & 15);
      const int tok = (dir == 0) ? ((b * RESL + j) * RESL + q)
                                 : ((b * RESL + q) * RESL + j);
#pragma unroll
      for (int v = 0; v < 8; ++v) {
        const int d2 = mt * 16 + v + ((lane >> 4) << 3);
        ymix[(size_t)tok * DIMC + head * 48 + d2] = (_Float16)acc[v];
      }
    }
  }
}

// ---------------------------------------------------------------- big GEMM
// out[M,384] = A[M,384] @ W[384,384]^T (+bias). One block = 64 rows, 8 waves.
template <bool AF32, bool OUTF32>
__global__ void __launch_bounds__(256)
gemm384_kernel(const void* __restrict__ Ap, const _Float16* __restrict__ W,
               const float* __restrict__ bias, void* __restrict__ outp) {
  __shared__ __align__(16) _Float16 xs[64][DIMC + 8];    // 64x392 f16 = 50176 B
  const int tid  = threadIdx.x;
  const int lane = tid & 31;
  const int wave = tid >> 5;
  const size_t row0 = (size_t)blockIdx.x * 64;

  if (AF32) {
    const float* A = (const float*)Ap;
    for (int i = tid; i < 64 * (DIMC / 4); i += 256) {
      const int r = i / (DIMC / 4), c4 = (i % (DIMC / 4)) * 4;
      const float4 f = *(const float4*)(A + (row0 + r) * DIMC + c4);
      xs[r][c4 + 0] = (_Float16)f.x;
      xs[r][c4 + 1] = (_Float16)f.y;
      xs[r][c4 + 2] = (_Float16)f.z;
      xs[r][c4 + 3] = (_Float16)f.w;
    }
  } else {
    // f16 A panel: pure byte copy -> async global->LDS path (ASYNCcnt)
    const _Float16* A = (const _Float16*)Ap;
    for (int i = tid; i < 64 * (DIMC / 8); i += 256) {
      const int r = i / (DIMC / 8), c8 = (i % (DIMC / 8)) * 8;
      cp16_g2l(A + (row0 + r) * DIMC + c8, &xs[r][c8]);
    }
    cp_fence_lds();
  }
  __syncthreads();

  const int mt  = wave >> 1;            // 4 M-tiles, 2 waves each
  const int nt0 = (wave & 1) * 12;      // each wave: 12 of 24 N-tiles
  for (int nn = 0; nn < 12; ++nn) {
    const int nt = nt0 + nn;
    // warm the next N-panel of W in L2/L0 (global_prefetch_b8)
    if (nn + 1 < 12)
      __builtin_prefetch(W + (size_t)((nt + 1) * 16) * DIMC, 0, 3);
    v8f acc = {};
#pragma unroll
    for (int k = 0; k < 12; ++k) {      // K = 384 = 12 x 32
      const v16h a = ldA(&xs[mt * 16][k * 32], DIMC + 8);
      const v16h b = ldBt(W + (size_t)(nt * 16) * DIMC + k * 32, DIMC);
      acc = wmma32(a, b, acc);
    }
    const int n = nt * 16 + (lane & 15);
    const float bv = bias ? bias[n] : 0.f;
#pragma unroll
    for (int v = 0; v < 8; ++v) {
      const size_t r = row0 + mt * 16 + v + ((lane >> 4) << 3);
      if (OUTF32) ((float*)outp)[r * DIMC + n] = acc[v] + bv;
      else ((_Float16*)outp)[r * DIMC + n] = (_Float16)(acc[v] + bv);
    }
  }
}

// ---------------------------------------------------------------- pooling
__global__ void __launch_bounds__(384)
pool_kernel(const _Float16* __restrict__ h, const _Float16* __restrict__ w,
            const _Float16* __restrict__ c, float* __restrict__ a) {
  const int b = blockIdx.x, ch = threadIdx.x;
  const size_t base = (size_t)b * (RESL * RESL) * DIMC + ch;
  float sum = 0.f;
  for (int t = 0; t < RESL * RESL; ++t) {
    const size_t idx = base + (size_t)t * DIMC;
    sum += (float)h[idx] + (float)w[idx] + (float)c[idx];
  }
  a[b * DIMC + ch] = sum * (1.f / (RESL * RESL));
}

// ---------------------------------------------------------------- reweighting
__global__ void __launch_bounds__(128)
reweight_kernel(const float* __restrict__ a, const float* __restrict__ Wr1,
                const float* __restrict__ br1, const float* __restrict__ Wr2,
                const float* __restrict__ br2, float* __restrict__ attn) {
  __shared__ float av[DIMC];
  __shared__ float r1[96];
  const int b = blockIdx.x, tid = threadIdx.x;
  for (int i = tid; i < DIMC; i += 128) av[i] = a[b * DIMC + i];
  __syncthreads();
  for (int o = tid; o < 96; o += 128) {
    float s = br1[o];
    for (int k = 0; k < DIMC; ++k) s += av[k] * Wr1[o * DIMC + k];
    r1[o] = 0.5f * s * (1.f + erff(s * 0.70710678118654752f));  // exact gelu
  }
  __syncthreads();
  for (int cch = tid; cch < DIMC; cch += 128) {
    float e[3];
    float m = -1e30f;
    for (int k3 = 0; k3 < 3; ++k3) {
      const int o = cch * 3 + k3;
      float s = br2[o];
      for (int j = 0; j < 96; ++j) s += r1[j] * Wr2[o * 96 + j];
      e[k3] = s;
      m = fmaxf(m, s);
    }
    float den = 0.f;
    for (int k3 = 0; k3 < 3; ++k3) { e[k3] = __expf(e[k3] - m); den += e[k3]; }
    for (int k3 = 0; k3 < 3; ++k3)
      attn[((size_t)b * DIMC + cch) * 3 + k3] = e[k3] / den;
  }
}

// ---------------------------------------------------------------- combine
__global__ void __launch_bounds__(256)
combine_kernel(const _Float16* __restrict__ h, const _Float16* __restrict__ w,
               const _Float16* __restrict__ c, const float* __restrict__ attn,
               _Float16* __restrict__ comb) {
  const size_t stride = (size_t)gridDim.x * blockDim.x;
  const size_t total = (size_t)NTOK * DIMC;
  for (size_t i = (size_t)blockIdx.x * blockDim.x + threadIdx.x; i < total;
       i += stride) {
    const int ch = (int)(i % DIMC);
    const int b = (int)(i / ((size_t)DIMC * RESL * RESL));
    const float* at = attn + ((size_t)b * DIMC + ch) * 3;
    const float v =
        (float)h[i] * at[0] + (float)w[i] * at[1] + (float)c[i] * at[2];
    comb[i] = (_Float16)v;
  }
}

// ---------------------------------------------------------------- launch
extern "C" void kernel_launch(void* const* d_in, const int* in_sizes, int n_in,
                              void* d_out, int out_size, void* d_ws,
                              size_t ws_size, hipStream_t stream) {
  const float* x     = (const float*)d_in[0];
  const float* Wc_h  = (const float*)d_in[1];
  const float* bc_h  = (const float*)d_in[2];
  const float* Wg_h  = (const float*)d_in[3];
  const float* bg_h  = (const float*)d_in[4];
  const float* Wo_h  = (const float*)d_in[5];
  const float* bo_h  = (const float*)d_in[6];
  const float* Wc_w  = (const float*)d_in[7];
  const float* bc_w  = (const float*)d_in[8];
  const float* Wg_w  = (const float*)d_in[9];
  const float* bg_w  = (const float*)d_in[10];
  const float* Wo_w  = (const float*)d_in[11];
  const float* bo_w  = (const float*)d_in[12];
  const float* Wmlpc = (const float*)d_in[13];
  const float* Wr1   = (const float*)d_in[14];
  const float* br1   = (const float*)d_in[15];
  const float* Wr2   = (const float*)d_in[16];
  const float* br2   = (const float*)d_in[17];
  const float* Wp    = (const float*)d_in[18];
  const float* bp    = (const float*)d_in[19];

  char* ws = (char*)d_ws;
  const size_t F16TOK = (size_t)NTOK * DIMC * sizeof(_Float16);  // 50331648
  _Float16* ymix    = (_Float16*)(ws);
  _Float16* hbuf    = (_Float16*)(ws + F16TOK);
  _Float16* wbuf    = (_Float16*)(ws + 2 * F16TOK);
  _Float16* cbuf    = (_Float16*)(ws + 3 * F16TOK);
  char* p = ws + 4 * F16TOK;
  _Float16* Wg_h16  = (_Float16*)p;  p += 1024 * 64 * 2;
  _Float16* Wg_w16  = (_Float16*)p;  p += 1024 * 64 * 2;
  _Float16* Wo_h16  = (_Float16*)p;  p += DIMC * DIMC * 2;
  _Float16* Wo_w16  = (_Float16*)p;  p += DIMC * DIMC * 2;
  _Float16* Wmlpc16 = (_Float16*)p;  p += DIMC * DIMC * 2;
  _Float16* Wp16    = (_Float16*)p;  p += DIMC * DIMC * 2;
  float* apool      = (float*)p;     p += NBATCH * DIMC * 4;
  float* attn       = (float*)p;     p += NBATCH * DIMC * 3 * 4;

  // 1) cast weights used by WMMA paths to f16 (L2-resident afterwards)
  const int nWg = 1024 * 64, nWsq = DIMC * DIMC;
  cast_f32_f16_kernel<<<(nWg + 255) / 256, 256, 0, stream>>>(Wg_h, Wg_h16, nWg);
  cast_f32_f16_kernel<<<(nWg + 255) / 256, 256, 0, stream>>>(Wg_w, Wg_w16, nWg);
  cast_f32_f16_kernel<<<(nWsq + 255) / 256, 256, 0, stream>>>(Wo_h, Wo_h16, nWsq);
  cast_f32_f16_kernel<<<(nWsq + 255) / 256, 256, 0, stream>>>(Wo_w, Wo_w16, nWsq);
  cast_f32_f16_kernel<<<(nWsq + 255) / 256, 256, 0, stream>>>(Wmlpc, Wmlpc16, nWsq);
  cast_f32_f16_kernel<<<(nWsq + 255) / 256, 256, 0, stream>>>(Wp, Wp16, nWsq);

  // 2) H-direction mix + Wo_h projection -> hbuf
  mix_kernel<<<NSEQ, 256, 0, stream>>>(x, Wc_h, bc_h, Wg_h16, bg_h, ymix, 0);
  gemm384_kernel<false, false><<<NTOK / 64, 256, 0, stream>>>(ymix, Wo_h16, bo_h, hbuf);

  // 3) W-direction mix + Wo_w projection -> wbuf
  mix_kernel<<<NSEQ, 256, 0, stream>>>(x, Wc_w, bc_w, Wg_w16, bg_w, ymix, 1);
  gemm384_kernel<false, false><<<NTOK / 64, 256, 0, stream>>>(ymix, Wo_w16, bo_w, wbuf);

  // 4) channel path: c = x @ Wmlpc^T (f32 A staged to f16, no bias)
  gemm384_kernel<true, false><<<NTOK / 64, 256, 0, stream>>>(x, Wmlpc16, nullptr, cbuf);

  // 5) global average pool of (h + w + c)
  pool_kernel<<<NBATCH, 384, 0, stream>>>(hbuf, wbuf, cbuf, apool);

  // 6) reweighting MLP + 3-way softmax
  reweight_kernel<<<NBATCH, 128, 0, stream>>>(apool, Wr1, br1, Wr2, br2, attn);

  // 7) combine h/w/c with per-(b,c) weights (reuse ymix as comb buffer)
  combine_kernel<<<4096, 256, 0, stream>>>(hbuf, wbuf, cbuf, attn, ymix);

  // 8) final projection: out = comb @ Wp^T + bp (f32 output)
  gemm384_kernel<false, true><<<NTOK / 64, 256, 0, stream>>>(ymix, Wp16, bp, d_out);
}
